// AKT_CORE_33002528702779
// MI455X (gfx1250) — matrix operand order repaired and checked by reading, not compiled, not run on hardware
//
#include <hip/hip_runtime.h>
#include <cstddef>
#include <cstdint>

// ---------------- problem constants ----------------
#define BB   16
#define SS   512
#define DD   512
#define HH   8
#define DKK  64
#define DFFF 2048
#define BSR  (BB*SS)          // 8192 rows
#define NEGF (-1e32f)

typedef __attribute__((ext_vector_type(16))) __bf16 bf16x16;
typedef __attribute__((ext_vector_type(8)))  float  floatx8;
typedef __attribute__((ext_vector_type(8)))  unsigned int uintx8;

// ---------------- wave helpers (wave32) ----------------
__device__ __forceinline__ float wave_max(float v) {
#pragma unroll
  for (int o = 16; o > 0; o >>= 1) v = fmaxf(v, __shfl_xor(v, o, 32));
  return v;
}
__device__ __forceinline__ float wave_sum(float v) {
#pragma unroll
  for (int o = 16; o > 0; o >>= 1) v += __shfl_xor(v, o, 32);
  return v;
}
__device__ __forceinline__ float softplusf(float x) {
  return (x > 20.f) ? x : log1pf(__expf(x));
}
// pack two f32 -> one dword of two bf16 (v_cvt_pk_bf16_f32)
__device__ __forceinline__ unsigned int pkbf(float a, float b) {
  union { __bf16 h[2]; unsigned int u; } x;
  x.h[0] = (__bf16)a; x.h[1] = (__bf16)b;
  return x.u;
}

// A-fragment from an LDS row: elements [h*8, h*8+8) and [16+h*8, ...)
// -> two 16-byte loads (ds_load_b128).
__device__ __forceinline__ bf16x16 ld_fragA(const __bf16* row, int half) {
  const unsigned int* p0 = (const unsigned int*)(row + half * 8);
  const unsigned int* p1 = (const unsigned int*)(row + 16 + half * 8);
  uintx8 u;
  u[0] = p0[0]; u[1] = p0[1]; u[2] = p0[2]; u[3] = p0[3];
  u[4] = p1[0]; u[5] = p1[1]; u[6] = p1[2]; u[7] = p1[3];
  return __builtin_bit_cast(bf16x16, u);
}
// B-fragment from a transposed LDS row: elements [h*16, h*16+16) contiguous.
__device__ __forceinline__ bf16x16 ld_fragB(const __bf16* row, int half) {
  const unsigned int* p0 = (const unsigned int*)(row + half * 16);
  uintx8 u;
#pragma unroll
  for (int i = 0; i < 8; ++i) u[i] = p0[i];
  return __builtin_bit_cast(bf16x16, u);
}
__device__ __forceinline__ void pack4(bf16x16& f, int o, float4 v) {
  f[o] = (__bf16)v.x; f[o + 1] = (__bf16)v.y;
  f[o + 2] = (__bf16)v.z; f[o + 3] = (__bf16)v.w;
}

// =====================================================================
// GEMM: C[M,N] = act(A[M,K] @ W[K,N] + bias[N] (+ resid[M,N]))
// 512 threads = 16 waves; block tile 128x128; each wave 2x2 WMMA tiles.
// MODE: 0 = plain, 1 = ReLU, 2 = +resid
// =====================================================================
template<int MODE>
__global__ __launch_bounds__(512) void gemm_kernel(
    const float* __restrict__ A, const float* __restrict__ W,
    const float* __restrict__ bias, const float* __restrict__ resid,
    float* __restrict__ C, int N, int K)
{
  __shared__ __bf16 lA[128][40];    // padded row (80B, 16B-multiple)
  __shared__ __bf16 lBt[128][40];   // W tile stored transposed: [n][k]
  const int tid  = threadIdx.x;
  const int lane = tid & 31;
  const int wave = __builtin_amdgcn_readfirstlane(tid >> 5);
  const int half = lane >> 4;
  const int bm = blockIdx.y, bn = blockIdx.x;
  const int mrow0 = (wave >> 2) * 32;
  const int ncol0 = (wave & 3) * 32;

  floatx8 acc[2][2] = {};
  const int nK = K >> 5;
  for (int kc = 0; kc < nK; ++kc) {
    // A block 128x32: float4 loads, packed bf16 b64 stores
#pragma unroll
    for (int i = 0; i < 2; ++i) {
      int idx = tid + i * 512;                 // 0..1023
      int r = idx >> 3, c4 = idx & 7;          // row, float4-slot
      float4 f = *(const float4*)&A[(size_t)(bm * 128 + r) * K + kc * 32 + c4 * 4];
      uint2 u; u.x = pkbf(f.x, f.y); u.y = pkbf(f.z, f.w);
      *(uint2*)&lA[r][c4 * 4] = u;             // 8B aligned (80r + 8c4)
    }
    // W block 32x128: float4 loads, transposed scalar b16 stores
#pragma unroll
    for (int i = 0; i < 2; ++i) {
      int idx = tid + i * 512;                 // 0..1023
      int r = idx >> 5, c4 = idx & 31;         // k-row, float4-slot over cols
      float4 f = *(const float4*)&W[(size_t)(kc * 32 + r) * N + bn * 128 + c4 * 4];
      lBt[c4 * 4 + 0][r] = (__bf16)f.x;
      lBt[c4 * 4 + 1][r] = (__bf16)f.y;
      lBt[c4 * 4 + 2][r] = (__bf16)f.z;
      lBt[c4 * 4 + 3][r] = (__bf16)f.w;
    }
    __syncthreads();

    bf16x16 af[2], bf[2];
#pragma unroll
    for (int rt = 0; rt < 2; ++rt)
      af[rt] = ld_fragA(&lA[mrow0 + rt * 16 + (lane & 15)][0], half);
#pragma unroll
    for (int ct = 0; ct < 2; ++ct)
      bf[ct] = ld_fragA(&lBt[ncol0 + ct * 16 + (lane & 15)][0], half);
#pragma unroll
    for (int rt = 0; rt < 2; ++rt)
#pragma unroll
      for (int ct = 0; ct < 2; ++ct)
        acc[rt][ct] = __builtin_amdgcn_wmma_f32_16x16x32_bf16(
            false, af[rt], false, bf[ct], (short)0, acc[rt][ct], false, false);
    __syncthreads();
  }

#pragma unroll
  for (int rt = 0; rt < 2; ++rt) {
#pragma unroll
    for (int ct = 0; ct < 2; ++ct) {
      const int n = bn * 128 + ncol0 + ct * 16 + (lane & 15);
      const float bi = bias[n];
#pragma unroll
      for (int r = 0; r < 8; ++r) {
        int m = bm * 128 + mrow0 + rt * 16 + (half << 3) + r;
        float v = acc[rt][ct][r] + bi;
        if (MODE == 2) v += resid[(size_t)m * N + n];
        if (MODE == 1) v = fmaxf(v, 0.f);
        C[(size_t)m * N + n] = v;
      }
    }
  }
}

// =====================================================================
// Attention with distance decay. One block per (b, h, 16-row block).
// qk: [BSR, DD] shared q/k projection (kq_same); vv: [BSR, DD] v projection.
// out: [BSR, DD] in concat-head layout (col = h*64 + d).
// =====================================================================
__global__ __launch_bounds__(512) void attn_kernel(
    const float* __restrict__ qk, const float* __restrict__ vv,
    const float* __restrict__ gam, float* __restrict__ out,
    int mask_flag, int zero_pad)
{
  __shared__ float  sc[16][SS];     // 32 KB scores
  __shared__ __bf16 at[16][SS];     // 16 KB attn (bf16)

  const int tid  = threadIdx.x;
  const int lane = tid & 31;
  const int wave = __builtin_amdgcn_readfirstlane(tid >> 5);  // scalar wave id
  const int half = lane >> 4;
  const int ib = blockIdx.x & 31;
  const int h  = (blockIdx.x >> 5) & 7;
  const int b  = blockIdx.x >> 8;
  const size_t rowbase = (size_t)b * SS * DD;
  const int colq = h * DKK;

  // ---------- Phase A: scores[16][512] = q @ k^T / 8 ----------
  // A(q) fragment depends only on the row block: build once per k-chunk.
  bf16x16 afrag[2];
  {
    const int iq = ib * 16 + (lane & 15);
    const float* qbase = qk + rowbase + (size_t)iq * DD + colq + (half << 3);
#pragma unroll
    for (int kc = 0; kc < 2; ++kc) {
      const float* qp = qbase + kc * 32;
      pack4(afrag[kc], 0,  *(const float4*)(qp));
      pack4(afrag[kc], 4,  *(const float4*)(qp + 4));
      pack4(afrag[kc], 8,  *(const float4*)(qp + 16));
      pack4(afrag[kc], 12, *(const float4*)(qp + 20));
    }
  }
  for (int t = 0; t < 2; ++t) {
    const int jt = wave + t * 16;          // scalar (wave-uniform)
    const int n  = jt * 16 + (lane & 15);
    if (jt > ib) {                         // fully above diagonal: zero fill
#pragma unroll
      for (int r = 0; r < 8; ++r) sc[(half << 3) + r][n] = 0.f;
    } else {
      floatx8 acc = {};
      const int jk = jt * 16 + (lane & 15);
      const float* kbase = qk + rowbase + (size_t)jk * DD + colq + (half << 3);
#pragma unroll
      for (int kc = 0; kc < 2; ++kc) {
        bf16x16 bfr;
        const float* kp = kbase + kc * 32;
        pack4(bfr, 0,  *(const float4*)(kp));
        pack4(bfr, 4,  *(const float4*)(kp + 4));
        pack4(bfr, 8,  *(const float4*)(kp + 16));
        pack4(bfr, 12, *(const float4*)(kp + 20));
        acc = __builtin_amdgcn_wmma_f32_16x16x32_bf16(false, afrag[kc], false, bfr,
                                                      (short)0, acc, false, false);
      }
#pragma unroll
      for (int r = 0; r < 8; ++r)
        sc[(half << 3) + r][n] = acc[r] * 0.125f;
    }
  }
  __syncthreads();

  // ---------- Phase B: per-row decay + double softmax (wave w -> row w) ----
  {
    const int m = wave;
    const int i = ib * 16 + m;
    const float g = -softplusf(gam[h]);
    const int j0 = lane * 16;

    float s[16], p[16];
    float vmax = -3.0e38f;
#pragma unroll
    for (int t = 0; t < 16; ++t) {
      int j = j0 + t;
      s[t] = sc[m][j];
      bool mk = mask_flag ? (j <= i) : (j < i);
      if (mk) vmax = fmaxf(vmax, s[t]);
    }
    vmax = wave_max(vmax);

    float lsum = 0.f;
#pragma unroll
    for (int t = 0; t < 16; ++t) {
      int j = j0 + t;
      bool mk = mask_flag ? (j <= i) : (j < i);
      p[t] = mk ? __expf(s[t] - vmax) : 0.f;
      lsum += p[t];
    }
    const float tot = wave_sum(lsum);
    const float inv = (tot > 0.f) ? (1.f / tot) : 0.f;

    // exclusive prefix of lane totals (shuffle scan)
    float incl = lsum;
#pragma unroll
    for (int off = 1; off < 32; off <<= 1) {
      float y = __shfl_up(incl, off, 32);
      if (lane >= off) incl += y;
    }
    float run = incl - lsum;               // exclusive prefix

#pragma unroll
    for (int t = 0; t < 16; ++t) {
      int j = j0 + t;
      run += p[t];                         // inclusive cumsum at j
      float suffix = (tot - run) * inv;    // sum_{j'>j} p (normalized)
      float pos = fabsf((float)(i - j));
      float dist = sqrtf(fmaxf(suffix * pos, 0.f));
      float te = fminf(fmaxf(__expf(dist * g), 1e-5f), 1e5f);
      bool mk = mask_flag ? (j <= i) : (j < i);
      s[t] = mk ? s[t] * te : NEGF;        // decayed scores
    }

    float m2 = -3.0e38f;
#pragma unroll
    for (int t = 0; t < 16; ++t) m2 = fmaxf(m2, s[t]);
    m2 = wave_max(m2);
    float ls2 = 0.f;
#pragma unroll
    for (int t = 0; t < 16; ++t) { p[t] = __expf(s[t] - m2); ls2 += p[t]; }
    const float inv2 = 1.f / wave_sum(ls2);
#pragma unroll
    for (int t = 0; t < 16; ++t) {
      float a = p[t] * inv2;
      if (zero_pad && i == 0) a = 0.f;     // attn[:, :, 0, :] = 0
      at[m][j0 + t] = (__bf16)a;
    }
  }
  __syncthreads();

  // ---------- Phase C: out[16][64] = attn @ v (4 waves, causal K skip) ----
  if (wave < 4) {                          // scalar branch: EXEC stays full
    const int nt = wave;
    const int col = colq + nt * 16 + (lane & 15);
    floatx8 acc = {};
    const int kcmax = ib >> 1;             // last j-chunk with unmasked cols
    for (int kc = 0; kc <= kcmax; ++kc) {
      bf16x16 a = ld_fragA(&at[lane & 15][kc * 32], half);
      bf16x16 bfr;
#pragma unroll
      for (int v = 0; v < 8; ++v) {
        int kb = (half << 3) + (v << 1) + ((v >> 2) << 3);
        const float* vp = vv + rowbase + (size_t)(kc * 32 + kb) * DD + col;
        bfr[2 * v]     = (__bf16)vp[0];
        bfr[2 * v + 1] = (__bf16)vp[DD];
      }
      acc = __builtin_amdgcn_wmma_f32_16x16x32_bf16(false, a, false, bfr,
                                                    (short)0, acc, false, false);
    }
#pragma unroll
    for (int r = 0; r < 8; ++r) {
      int i = ib * 16 + (half << 3) + r;
      out[rowbase + (size_t)i * DD + col] = acc[r];
    }
  }
}

// =====================================================================
// LayerNorm: one 512-wide row per 256-thread block.
// =====================================================================
__device__ __forceinline__ float block_sum256(float v, float* red) {
  v = wave_sum(v);
  const int lane = threadIdx.x & 31, w = threadIdx.x >> 5;
  if (lane == 0) red[w] = v;
  __syncthreads();
  float t = (lane < 8) ? red[lane] : 0.f;
  t = wave_sum(t);
  __syncthreads();
  return t;
}

__global__ __launch_bounds__(256) void ln_kernel(
    const float* __restrict__ x, const float* __restrict__ g,
    const float* __restrict__ be, float* __restrict__ y)
{
  __shared__ float red[8];
  const size_t base = (size_t)blockIdx.x * DD;
  const int t = threadIdx.x;
  float a = x[base + t], b = x[base + t + 256];
  float mu = block_sum256(a + b, red) * (1.f / 512.f);
  float da = a - mu, db = b - mu;
  float var = block_sum256(da * da + db * db, red) * (1.f / 512.f);
  float inv = rsqrtf(var + 1e-5f);
  y[base + t]       = da * inv * g[t]       + be[t];
  y[base + t + 256] = db * inv * g[t + 256] + be[t + 256];
}

// =====================================================================
// Host orchestration
// =====================================================================
static inline void gemm(hipStream_t st, const float* A, const float* W,
                        const float* bias, const float* resid, float* C,
                        int M, int N, int K, int mode) {
  dim3 grid(N / 128, M / 128);
  if (mode == 2)
    gemm_kernel<2><<<grid, 512, 0, st>>>(A, W, bias, resid, C, N, K);
  else if (mode == 1)
    gemm_kernel<1><<<grid, 512, 0, st>>>(A, W, bias, nullptr, C, N, K);
  else
    gemm_kernel<0><<<grid, 512, 0, st>>>(A, W, bias, nullptr, C, N, K);
}

struct WS { float *qk, *vv, *att, *s1, *q1, *h1; };

static void run_layer(hipStream_t st, int l, const float* qsrc, const float* vsrc,
                      float* dst, int mask_flag, int apply_pos,
                      void* const* d_in, const WS& w)
{
  const float* Wk  = (const float*)d_in[3]  + (size_t)l * DD * DD;
  const float* bk  = (const float*)d_in[4]  + (size_t)l * DD;
  const float* Wv  = (const float*)d_in[5]  + (size_t)l * DD * DD;
  const float* bv  = (const float*)d_in[6]  + (size_t)l * DD;
  const float* Wo  = (const float*)d_in[7]  + (size_t)l * DD * DD;
  const float* bo  = (const float*)d_in[8]  + (size_t)l * DD;
  const float* gam = (const float*)d_in[9]  + (size_t)l * HH;
  const float* g1  = (const float*)d_in[10] + (size_t)l * DD;
  const float* b1n = (const float*)d_in[11] + (size_t)l * DD;
  const float* W1  = (const float*)d_in[12] + (size_t)l * DD * DFFF;
  const float* bf1 = (const float*)d_in[13] + (size_t)l * DFFF;
  const float* W2  = (const float*)d_in[14] + (size_t)l * DFFF * DD;
  const float* bf2 = (const float*)d_in[15] + (size_t)l * DD;
  const float* g2  = (const float*)d_in[16] + (size_t)l * DD;
  const float* b2n = (const float*)d_in[17] + (size_t)l * DD;

  // q/k share Wk and the same source tensor (kq_same=True, q==k in all calls)
  gemm(st, qsrc, Wk, bk, nullptr, w.qk, BSR, DD, DD, 0);
  gemm(st, vsrc, Wv, bv, nullptr, w.vv, BSR, DD, DD, 0);
  attn_kernel<<<BB * HH * (SS / 16), 512, 0, st>>>(
      w.qk, w.vv, gam, w.att, mask_flag, mask_flag == 0 ? 1 : 0);
  gemm(st, w.att, Wo, bo, qsrc, w.s1, BSR, DD, DD, 2);   // + residual(query)
  ln_kernel<<<BSR, 256, 0, st>>>(w.s1, g1, b1n, apply_pos ? w.q1 : dst);

  if (apply_pos) {
    for (int c = 0; c < 4; ++c) {                        // FFN in 2048-row chunks
      const float* Ain = w.q1 + (size_t)c * 2048 * DD;
      float*       Cch = w.s1 + (size_t)c * 2048 * DD;
      gemm(st, Ain, W1, bf1, nullptr, w.h1, 2048, DFFF, DD, 1);   // ReLU
      gemm(st, w.h1, W2, bf2, Ain, Cch, 2048, DD, DFFF, 2);       // + residual
    }
    ln_kernel<<<BSR, 256, 0, st>>>(w.s1, g2, b2n, dst);
  }
}

extern "C" void kernel_launch(void* const* d_in, const int* in_sizes, int n_in,
                              void* d_out, int out_size, void* d_ws, size_t ws_size,
                              hipStream_t stream)
{
  (void)in_sizes; (void)n_in; (void)out_size;
  const size_t T = (size_t)BSR * DD;                     // 4,194,304 floats
  const size_t need = 6 * T * sizeof(float);             // ~96 MB
  if (ws_size < need) return;                            // cannot proceed safely

  WS w;
  w.qk  = (float*)d_ws;
  w.vv  = w.qk  + T;
  w.att = w.vv  + T;
  w.s1  = w.att + T;
  w.q1  = w.s1  + T;
  w.h1  = w.q1  + T;                                     // 2048x2048 chunk

  const float* q_emb  = (const float*)d_in[0];
  const float* qa_emb = (const float*)d_in[1];
  float* xout = (float*)d_out;                           // tuple (x, y): x first
  float* yout = xout + T;

  // blocks_1: y chain (self-attn, mask=1, FFN)
  run_layer(stream, 0, qa_emb, qa_emb, yout, 1, 1, d_in, w);
  run_layer(stream, 1, yout,   yout,   yout, 1, 1, d_in, w);
  // blocks_2: x chain, alternating (mask=1, no FFN) / (mask=0, values=y, FFN)
  run_layer(stream, 2, q_emb, q_emb, xout, 1, 0, d_in, w);
  run_layer(stream, 3, xout,  yout,  xout, 0, 1, d_in, w);
  run_layer(stream, 4, xout,  xout,  xout, 1, 0, d_in, w);
  run_layer(stream, 5, xout,  yout,  xout, 0, 1, d_in, w);
}
